// MultiheadGQA_38611755991280
// MI455X (gfx1250) — matrix operand order repaired and checked by reading, not compiled, not run on hardware
//
#include <hip/hip_runtime.h>
#include <hip/hip_bf16.h>

typedef __bf16 bf16_t;
typedef __attribute__((ext_vector_type(16))) __bf16 v16bf;
typedef __attribute__((ext_vector_type(8)))  __bf16 v8bf;
typedef __attribute__((ext_vector_type(4)))  __bf16 v4bf;
typedef __attribute__((ext_vector_type(8)))  float  v8f;

#define E_DIM   2048
#define HQ_N    16
#define HK_N    4
#define D_HEAD  128
#define G_GRP   4
#define KVE_DIM 512
#define B_SZ    2
#define N_SEQ   2048

// ---------------------------------------------------------------------------
// WMMA helpers (bf16 in, f32 accumulate)
// ---------------------------------------------------------------------------
__device__ __forceinline__ v8f wmma_bf16(v16bf a, v16bf b, v8f c) {
    // D = A(16x32) * B(32x16) + C(16x16 f32)
    return __builtin_amdgcn_wmma_f32_16x16x32_bf16(
        false, a, false, b, (short)0, c, false, false);
}

// A-fragment (16 rows x 32 K, 16-bit): lane l holds row l%16;
// lane-half 0 -> K {0..15}, lane-half 1 -> K {16..31}
__device__ __forceinline__ v16bf load_frag_a(const bf16_t* __restrict__ base,
                                             int stride, int lane) {
    const int row = lane & 15;
    const int off = ((lane >> 4) & 1) * 16;
    const bf16_t* p = base + (size_t)row * stride + off;
    v8bf lo = *(const v8bf*)p;          // K off..off+7
    v8bf hi = *(const v8bf*)(p + 8);    // K off+8..off+15
    return __builtin_shufflevector(lo, hi, 0,1,2,3,4,5,6,7,8,9,10,11,12,13,14,15);
}

// B-fragment (32 K x 16 cols, 16-bit): lane l holds column l%16;
// lane-half selects contiguous K half (16 values)
__device__ __forceinline__ v16bf load_frag_b(const bf16_t* __restrict__ base,
                                             int stride, int lane) {
    const int col = lane & 15;
    const int off = ((lane >> 4) & 1) * 16;
    return *(const v16bf*)(base + (size_t)col * stride + off);
}

__device__ __forceinline__ unsigned pack2_bf16(float a, float b) {
    union { bf16_t h; unsigned short s; } x, y;
    x.h = (bf16_t)a; y.h = (bf16_t)b;
    return (unsigned)x.s | ((unsigned)y.s << 16);
}

// ---------------------------------------------------------------------------
// f32 -> bf16 conversion (4 elems / thread)
// ---------------------------------------------------------------------------
__global__ __launch_bounds__(256) void cvt_f32_bf16(const float* __restrict__ in,
                                                    bf16_t* __restrict__ out, int n4) {
    int i = blockIdx.x * 256 + threadIdx.x;
    if (i >= n4) return;
    float4 v = ((const float4*)in)[i];
    v4bf t;
    t.x = (bf16_t)v.x; t.y = (bf16_t)v.y; t.z = (bf16_t)v.z; t.w = (bf16_t)v.w;
    *(v4bf*)(out + 4 * (size_t)i) = t;
}

// ---------------------------------------------------------------------------
// V transpose: Vp[b][s][hk*D+d] -> Vt[((b*HK+hk)*D+d)*N + s]
// ---------------------------------------------------------------------------
__global__ __launch_bounds__(256) void transpose_v(const bf16_t* __restrict__ vp,
                                                   bf16_t* __restrict__ vt) {
    int idx = blockIdx.x * 256 + threadIdx.x;          // B*HK*D*N total
    int s  = idx & (N_SEQ - 1);
    int t  = idx >> 11;                                // /N_SEQ
    int d  = t & (D_HEAD - 1);
    t >>= 7;                                           // /D_HEAD
    int hk = t & (HK_N - 1);
    int b  = t >> 2;
    vt[idx] = vp[((size_t)(b * N_SEQ + s)) * KVE_DIM + hk * D_HEAD + d];
}

// ---------------------------------------------------------------------------
// GEMM: C[M x Nout] = (A[M x K] @ W[Nout x K]^T + bias) * scale
// (torch Linear convention). Wave computes 32(M) x 64(N); block 128x128.
// ---------------------------------------------------------------------------
__global__ __launch_bounds__(256, 2) void gemm_bf16(
        const bf16_t* __restrict__ A, const bf16_t* __restrict__ W,
        const float* __restrict__ bias, bf16_t* __restrict__ outB,
        float* __restrict__ outF, int M, int Nout, int K, float scale) {
    const int lane = threadIdx.x & 31;
    const int wave = threadIdx.x >> 5;
    const int m0 = blockIdx.y * 128 + (wave & 3) * 32;
    const int n0 = blockIdx.x * 128 + (wave >> 2) * 64;

    v8f acc[2][4];
#pragma unroll
    for (int i = 0; i < 2; ++i)
#pragma unroll
        for (int j = 0; j < 4; ++j)
#pragma unroll
            for (int e = 0; e < 8; ++e) acc[i][j][e] = 0.f;

    const bf16_t* a0p = A + (size_t)m0 * K;
    const bf16_t* a1p = A + (size_t)(m0 + 16) * K;
    for (int kk = 0; kk < K; kk += 32) {
        v16bf a0 = load_frag_a(a0p + kk, K, lane);
        v16bf a1 = load_frag_a(a1p + kk, K, lane);
#pragma unroll
        for (int j = 0; j < 4; ++j) {
            v16bf b = load_frag_b(W + (size_t)(n0 + j * 16) * K + kk, K, lane);
            acc[0][j] = wmma_bf16(a0, b, acc[0][j]);
            acc[1][j] = wmma_bf16(a1, b, acc[1][j]);
        }
    }

    const int lh  = (lane >> 4) & 1;
    const int col = lane & 15;
#pragma unroll
    for (int j = 0; j < 4; ++j) {
        const int n = n0 + j * 16 + col;
        const float bv = bias ? bias[n] : 0.f;
#pragma unroll
        for (int i = 0; i < 2; ++i) {
#pragma unroll
            for (int r = 0; r < 8; ++r) {
                const int row = m0 + i * 16 + lh * 8 + r;
                const float v = (acc[i][j][r] + bv) * scale;
                if (outF) outF[(size_t)row * Nout + n] = v;
                else      outB[(size_t)row * Nout + n] = (bf16_t)v;
            }
        }
    }
}

// ---------------------------------------------------------------------------
// Flash attention (GQA). One wave owns 16 query rows of one (b, hq) head.
// Computes S^T = K @ Q^T so softmax reductions are in-lane, then O^T = V^T @ P.
// ---------------------------------------------------------------------------
__global__ __launch_bounds__(256, 2) void gqa_attn(
        const bf16_t* __restrict__ Qp, const bf16_t* __restrict__ Kp,
        const bf16_t* __restrict__ Vt, bf16_t* __restrict__ AO,
        const int* __restrict__ causal_p) {
    const int causal = *causal_p;
    const int lane = threadIdx.x & 31;
    const int wave = threadIdx.x >> 5;
    const int w    = blockIdx.x * 8 + wave;     // 0 .. B*HQ*(N/16)-1
    const int qblk = w & (N_SEQ / 16 - 1);      // 0..127
    const int head = w >> 7;                    // 0..B*HQ-1
    const int hq   = head & (HQ_N - 1);
    const int b    = head >> 4;
    const int hk   = hq >> 2;                   // hq / G
    const int n0   = qblk * 16;
    const int lh   = (lane >> 4) & 1;
    const int q    = lane & 15;

    const bf16_t* Qb = Qp + ((size_t)(b * N_SEQ + n0)) * E_DIM + hq * D_HEAD;
    const bf16_t* Kb = Kp + ((size_t)(b * N_SEQ)) * KVE_DIM + hk * D_HEAD;
    const bf16_t* Vb = Vt + ((size_t)(b * HK_N + hk)) * D_HEAD * N_SEQ;

    // hoist Q^T fragments (whole 16x128 tile, 32 VGPRs)
    v16bf bq[4];
#pragma unroll
    for (int d0 = 0; d0 < 4; ++d0) bq[d0] = load_frag_b(Qb + d0 * 32, E_DIM, lane);

    v8f acc[8];
#pragma unroll
    for (int t = 0; t < 8; ++t)
#pragma unroll
        for (int e = 0; e < 8; ++e) acc[t][e] = 0.f;

    float m_run = -1e30f, l_run = 0.f;
    const int kend = causal ? (n0 + 16) : N_SEQ;

    for (int kb = 0; kb < kend; kb += 32) {
        // S^T tiles: keys kb..kb+15 (st0) and kb+16..kb+31 (st1), queries n0..n0+15
        v8f st0, st1;
#pragma unroll
        for (int e = 0; e < 8; ++e) { st0[e] = 0.f; st1[e] = 0.f; }
#pragma unroll
        for (int d0 = 0; d0 < 4; ++d0) {
            v16bf k0 = load_frag_a(Kb + (size_t)kb * KVE_DIM + d0 * 32, KVE_DIM, lane);
            v16bf k1 = load_frag_a(Kb + (size_t)(kb + 16) * KVE_DIM + d0 * 32, KVE_DIM, lane);
            st0 = wmma_bf16(k0, bq[d0], st0);
            st1 = wmma_bf16(k1, bq[d0], st1);
        }

        if (causal && (kb + 31 > n0)) {
            const int qa = n0 + q;
#pragma unroll
            for (int r = 0; r < 8; ++r) {
                const int ka = kb + lh * 8 + r;
                if (ka > qa)      st0[r] = -1e30f;
                if (ka + 16 > qa) st1[r] = -1e30f;
            }
        }

        // online softmax; partner lane (lane^16) holds same query, other keys
        float mb = -1e30f;
#pragma unroll
        for (int r = 0; r < 8; ++r) mb = fmaxf(mb, fmaxf(st0[r], st1[r]));
        mb = fmaxf(mb, __shfl_xor(mb, 16, 32));
        const float m_new = fmaxf(m_run, mb);
        const float corr  = __expf(m_run - m_new);
        float rs = 0.f;
#pragma unroll
        for (int r = 0; r < 8; ++r) {
            st0[r] = __expf(st0[r] - m_new); rs += st0[r];
            st1[r] = __expf(st1[r] - m_new); rs += st1[r];
        }
        rs += __shfl_xor(rs, 16, 32);
        l_run = l_run * corr + rs;
        m_run = m_new;
#pragma unroll
        for (int t = 0; t < 8; ++t)
#pragma unroll
            for (int e = 0; e < 8; ++e) acc[t][e] *= corr;

        // build P B-fragment (K=32 keys, N=16 queries) from S^T C-layout
        unsigned own0[4], own1[4], oth0[4], oth1[4];
#pragma unroll
        for (int j = 0; j < 4; ++j) {
            own0[j] = pack2_bf16(st0[2 * j], st0[2 * j + 1]);
            own1[j] = pack2_bf16(st1[2 * j], st1[2 * j + 1]);
        }
#pragma unroll
        for (int j = 0; j < 4; ++j) {
            oth0[j] = __shfl_xor(own0[j], 16, 32);
            oth1[j] = __shfl_xor(own1[j], 16, 32);
        }
        union { v16bf v; unsigned u[8]; } pf;
#pragma unroll
        for (int j = 0; j < 4; ++j) {
            pf.u[j]     = lh ? oth1[j] : own0[j];   // keys lh*16 + 0..7
            pf.u[4 + j] = lh ? own1[j] : oth0[j];   // keys lh*16 + 8..15
        }

        // O^T += V^T(d-tile, keys) @ P
#pragma unroll
        for (int t = 0; t < 8; ++t) {
            v16bf av = load_frag_a(Vb + (size_t)(t * 16) * N_SEQ + kb, N_SEQ, lane);
            acc[t] = wmma_bf16(av, pf.v, acc[t]);
        }
    }

    const float inv = 1.f / l_run;
    bf16_t* Ob = AO + ((size_t)(b * N_SEQ + n0 + q)) * E_DIM + hq * D_HEAD + lh * 8;
#pragma unroll
    for (int t = 0; t < 8; ++t) {
        union { v8bf v; bf16_t h[8]; } o;
#pragma unroll
        for (int r = 0; r < 8; ++r) o.h[r] = (bf16_t)(acc[t][r] * inv);
        *(v8bf*)(Ob + t * 16) = o.v;
    }
}

// ---------------------------------------------------------------------------
// Host side
// ---------------------------------------------------------------------------
extern "C" void kernel_launch(void* const* d_in, const int* in_sizes, int n_in,
                              void* d_out, int out_size, void* d_ws, size_t ws_size,
                              hipStream_t stream) {
    const float* q_in = (const float*)d_in[0];
    const float* k_in = (const float*)d_in[1];
    const float* v_in = (const float*)d_in[2];
    const float* Wq   = (const float*)d_in[3];
    const float* bq   = (const float*)d_in[4];
    const float* Wk   = (const float*)d_in[5];
    const float* bk   = (const float*)d_in[6];
    const float* Wv   = (const float*)d_in[7];
    const float* bv   = (const float*)d_in[8];
    const float* Wo   = (const float*)d_in[9];
    const float* bo   = (const float*)d_in[10];
    const int*   isc  = (const int*)d_in[11];
    float* out = (float*)d_out;

    const size_t SZ_IN = (size_t)B_SZ * N_SEQ * E_DIM;    // 8.39M
    const size_t SZ_WQ = (size_t)E_DIM * E_DIM;           // 4.19M
    const size_t SZ_WK = (size_t)KVE_DIM * E_DIM;         // 1.05M
    const size_t SZ_KV = (size_t)B_SZ * N_SEQ * KVE_DIM;  // 2.10M

    bf16_t* ws = (bf16_t*)d_ws;
    size_t off = 0;
    bf16_t* qx = ws + off; off += SZ_IN;
    bf16_t* kx = ws + off; off += SZ_IN;
    bf16_t* vx = ws + off; off += SZ_IN;
    bf16_t* wq = ws + off; off += SZ_WQ;
    bf16_t* wk = ws + off; off += SZ_WK;
    bf16_t* wv = ws + off; off += SZ_WK;
    bf16_t* wo = ws + off; off += SZ_WQ;
    bf16_t* Qp = ws + off; off += SZ_IN;
    bf16_t* Kp = ws + off; off += SZ_KV;
    bf16_t* Vp = ws + off; off += SZ_KV;
    bf16_t* Vt = kx;   // kx dead after K projection
    bf16_t* AO = qx;   // qx dead after Q projection

    const int M = B_SZ * N_SEQ;  // 4096
    const float qscale = 1.0f / sqrtf((float)D_HEAD);

    // 1) f32 -> bf16
    auto cvt = [&](const float* src, bf16_t* dst, size_t n) {
        int n4 = (int)(n / 4);
        cvt_f32_bf16<<<(n4 + 255) / 256, 256, 0, stream>>>(src, dst, n4);
    };
    cvt(q_in, qx, SZ_IN);  cvt(k_in, kx, SZ_IN);  cvt(v_in, vx, SZ_IN);
    cvt(Wq, wq, SZ_WQ);    cvt(Wk, wk, SZ_WK);
    cvt(Wv, wv, SZ_WK);    cvt(Wo, wo, SZ_WQ);

    // 2) projections (bf16 out); Q pre-scaled by 1/sqrt(D)
    gemm_bf16<<<dim3(E_DIM / 128, M / 128), 256, 0, stream>>>(
        qx, wq, bq, Qp, nullptr, M, E_DIM, E_DIM, qscale);
    gemm_bf16<<<dim3(KVE_DIM / 128, M / 128), 256, 0, stream>>>(
        kx, wk, bk, Kp, nullptr, M, KVE_DIM, E_DIM, 1.0f);
    gemm_bf16<<<dim3(KVE_DIM / 128, M / 128), 256, 0, stream>>>(
        vx, wv, bv, Vp, nullptr, M, KVE_DIM, E_DIM, 1.0f);

    // 3) V transpose -> [b][hk][d][s]
    transpose_v<<<(int)(SZ_KV / 256), 256, 0, stream>>>(Vp, Vt);

    // 4) flash attention: 1 wave per 16 queries per head
    const int nwaves = B_SZ * HQ_N * (N_SEQ / 16);        // 4096
    gqa_attn<<<nwaves / 8, 256, 0, stream>>>(Qp, Kp, Vt, AO, isc);

    // 5) output projection -> f32 d_out
    gemm_bf16<<<dim3(E_DIM / 128, M / 128), 256, 0, stream>>>(
        AO, wo, bo, nullptr, out, M, E_DIM, E_DIM, 1.0f);

    (void)in_sizes; (void)n_in; (void)out_size; (void)ws_size;
}
